// LSTM_CRF_28862180229383
// MI455X (gfx1250) — matrix operand order repaired and checked by reading, not compiled, not run on hardware
//
#include <hip/hip_runtime.h>
#include <cstddef>
#include <cstdint>

// Problem constants (from reference): B=128, T=512, K=128
#define TT  512
#define KK  128
#define BB  128
#define BPG 4            // batches per workgroup -> 32 workgroups
#define NTHR 256         // 8 wave32s: one per 16-column WMMA tile

typedef __attribute__((ext_vector_type(16))) _Float16 v16h;
typedef __attribute__((ext_vector_type(8)))  _Float16 v8h;
typedef __attribute__((ext_vector_type(8)))  float    v8f;

// Dynamic-LDS layout (bytes):
//   trL    [128][128] f32 : 0      .. 65536
//   alpha  [BPG][128] f32 : 65536  .. 67584
//   svit   [BPG][128] f32 : 67584  .. 69632
//   ptile  [16][128]  f16 : 69632  .. 73728   (rows BPG..15 stay zero)
//   rowmax [BPG]      f32 : 73728  .. 73744
#define SMEM_BYTES 73744

// ---------------------------------------------------------------------------
// Kernel 1: Et[j][i] = exp(transitions[i][j])  (f16, transposed so that each
// lane's WMMA B-fragment is one contiguous 32-byte load)
// ---------------------------------------------------------------------------
__global__ void crf_prep_et(const float* __restrict__ tr,
                            _Float16* __restrict__ Et) {
    int j = blockIdx.x;   // column of transitions
    int i = threadIdx.x;  // row of transitions
    Et[j * KK + i] = (_Float16)__expf(tr[i * KK + j]);
}

// ---------------------------------------------------------------------------
// Kernel 2: gold-path score per batch (one wave32 per batch)
// ---------------------------------------------------------------------------
__global__ void crf_score(const float* __restrict__ em,
                          const float* __restrict__ tr,
                          const float* __restrict__ startv,
                          const float* __restrict__ endv,
                          const int*   __restrict__ tags,
                          const int*   __restrict__ seq_lens,
                          float* __restrict__ score) {
    const int b    = blockIdx.x;
    const int lane = threadIdx.x;      // 0..31
    const int len  = seq_lens[b] + 1;  // lengths = seq_lens + 1

    float acc = 0.0f;
    for (int t = 1 + lane; t < TT; t += 32) {
        if (t < len) {
            int pt = tags[b * TT + t - 1];
            int ct = tags[b * TT + t];
            acc += tr[pt * KK + ct] + em[((size_t)b * TT + t) * KK + ct];
        }
    }
    #pragma unroll
    for (int off = 16; off > 0; off >>= 1)
        acc += __shfl_xor(acc, off, 32);

    if (lane == 0) {
        int t0 = tags[b * TT + 0];
        int lt = tags[b * TT + (len - 1)];
        score[b] = acc + startv[t0] + em[((size_t)b * TT + 0) * KK + t0] + endv[lt];
    }
}

// ---------------------------------------------------------------------------
// Kernel 3: the sequential scan.  32 workgroups x 256 threads (8 wave32s).
//   wave w owns WMMA column tile j0 = w*16 (exp-domain forward recurrence
//   via v_wmma_f32_16x16x32_f16); all waves share Viterbi (2 columns/thread,
//   transitions LDS-resident).
// ---------------------------------------------------------------------------
__global__ __launch_bounds__(NTHR)
void crf_scan(const float* __restrict__ em,
              const float* __restrict__ tr,
              const float* __restrict__ startv,
              const float* __restrict__ endv,
              const int*   __restrict__ seq_lens,
              const _Float16* __restrict__ Et,
              float* __restrict__ logz,
              int*   __restrict__ last_tag,
              unsigned char* __restrict__ bp) {
    extern __shared__ __align__(32) char smem[];
    float*    trL    = (float*)(smem);
    float*    alpha  = (float*)(smem + 65536);
    float*    svit   = (float*)(smem + 67584);
    _Float16* ptile  = (_Float16*)(smem + 69632);   // 16 rows x 128
    float*    rowmax = (float*)(smem + 73728);

    const int tid  = threadIdx.x;
    const int wave = tid >> 5;   // 0..7
    const int lane = tid & 31;
    const int b0   = blockIdx.x * BPG;

    // --- loop-invariant lengths in registers (avoids per-step LDS loads) ---
    int lenR[BPG];
    #pragma unroll
    for (int r = 0; r < BPG; ++r) lenR[r] = seq_lens[b0 + r] + 1;

    // --- stage transitions into LDS (row-major, same as global) ---
    for (int u = tid; u < (KK * KK) / 4; u += NTHR)
        *(float4*)&trL[u * 4] = *(const float4*)&tr[u * 4];

    // --- init alpha0 = start + emissions[:,0,:]; Viterbi state = same ---
    for (int u = tid; u < BPG * KK; u += NTHR) {
        int b = u >> 7, j = u & 127;
        float v = startv[j] + em[((size_t)(b0 + b) * TT + 0) * KK + j];
        alpha[b * KK + j] = v;
        svit [b * KK + j] = v;
    }
    // zero all 16 ptile rows once; only rows 0..BPG-1 are rewritten per step,
    // rows BPG..15 contribute exact zeros to the WMMA.
    for (int u = tid; u < 16 * KK; u += NTHR) ptile[u] = (_Float16)0.0f;

    // --- preload the 4 B-fragments of Et for this wave's 16-column tile ---
    // B as this code loads it (contiguous k, same as A): lanes 0-15 hold K=0..15 (N=lane),
    //                           lanes 16-31 hold K=16..31 (N=lane-16).
    const int n = lane & 15;
    v16h bfrag[4];
    {
        const int j0   = wave * 16;
        const int koff = (lane < 16) ? 0 : 16;
        #pragma unroll
        for (int kb = 0; kb < 4; ++kb)
            bfrag[kb] = *(const v16h*)(Et + (j0 + n) * KK + kb * 32 + koff);
    }

    // Viterbi ownership: thread -> (vb, columns vj, vj+1)
    const int vb    = tid >> 6;          // 0..3
    const int vj    = (tid & 63) * 2;    // 0,2,..,126
    const int mylen = lenR[vb];

    __syncthreads();

    for (int t = 1; t < TT; ++t) {
        // ---- phase 1: per-row max of alpha (wave w<BPG reduces row w) ----
        if (wave < BPG) {
            float m = -1e30f;
            for (int j = lane; j < KK; j += 32) m = fmaxf(m, alpha[wave * KK + j]);
            #pragma unroll
            for (int off = 16; off > 0; off >>= 1)
                m = fmaxf(m, __shfl_xor(m, off, 32));
            if (lane == 0) rowmax[wave] = m;
        }
        __syncthreads();

        // ---- phase 2: p = exp(alpha - rowmax) as f16 (rows 0..BPG-1) ----
        for (int u = tid; u < BPG * KK; u += NTHR) {
            int b = u >> 7, j = u & 127;
            ptile[b * KK + j] = (_Float16)__expf(alpha[b * KK + j] - rowmax[b]);
        }
        __syncthreads();

        // ---- phase 3a: WMMA  C[b,j] = sum_i p[b,i] * E[i,j]  (all 8 waves) ----
        // A as this code loads it (contiguous k, same as B): lanes 0-15: M=lane,   K=0..15
        //                           lanes16-31: M=lane-16, K=16..31
        v8f cacc = {0.f, 0.f, 0.f, 0.f, 0.f, 0.f, 0.f, 0.f};
        {
            const int kbase = (lane < 16) ? 0 : 16;
            #pragma unroll
            for (int kb = 0; kb < 4; ++kb) {
                union { v16h v; v8h h[2]; } af;
                const _Float16* row = &ptile[n * KK + kb * 32 + kbase];
                af.h[0] = *(const v8h*)(row);
                af.h[1] = *(const v8h*)(row + 8);
                cacc = __builtin_amdgcn_wmma_f32_16x16x32_f16(
                    false, af.v, false, bfrag[kb], (short)0, cacc, false, false);
            }
        }

        // ---- phase 3b: Viterbi max-plus + argmax (all waves, LDS-resident) ----
        float best0 = -1e30f, best1 = -1e30f;
        int   bi0 = 0, bi1 = 0;
        for (int i = 0; i < KK; ++i) {
            float  sv  = svit[vb * KK + i];                 // LDS broadcast
            float2 trr = *(const float2*)&trL[i * KK + vj]; // LDS, conflict-free
            float c0 = sv + trr.x; if (c0 > best0) { best0 = c0; bi0 = i; }
            float c1 = sv + trr.y; if (c1 > best1) { best1 = c1; bi1 = i; }
        }

        // prefetch next step's emissions while we still compute
        if (t + 1 < TT)
            __builtin_prefetch(&em[((size_t)(b0 + vb) * TT + t + 1) * KK + vj], 0, 1);

        __syncthreads();

        // ---- phase 4: masked write-back ----
        // C/D layout: lanes 0-15: VGPR r -> M=r; only M=0..BPG-1 are live.
        if (lane < 16) {
            const int j = wave * 16 + n;
            #pragma unroll
            for (int r = 0; r < BPG; ++r) {
                if (t < lenR[r]) {
                    float nxt = rowmax[r] + __logf(cacc[r]) +
                                em[((size_t)(b0 + r) * TT + t) * KK + j];
                    alpha[r * KK + j] = nxt;
                }
            }
        }
        {
            const bool on = (t < mylen);
            uchar2 bpv;
            if (on) {
                const float* emr = &em[((size_t)(b0 + vb) * TT + t) * KK + vj];
                svit[vb * KK + vj + 0] = best0 + emr[0];
                svit[vb * KK + vj + 1] = best1 + emr[1];
                bpv = make_uchar2((unsigned char)bi0, (unsigned char)bi1);
            } else {
                bpv = make_uchar2((unsigned char)(vj + 0), (unsigned char)(vj + 1));
            }
            *(uchar2*)&bp[((size_t)(b0 + vb) * (TT - 1) + (t - 1)) * KK + vj] = bpv;
        }
        __syncthreads();
    }

    // ---- finalize: log_z and last_tag (wave w<BPG handles row w) ----
    if (wave < BPG) {
        const int b = wave;
        // logsumexp_j(alpha + end)
        float m = -1e30f;
        for (int j = lane; j < KK; j += 32) m = fmaxf(m, alpha[b * KK + j] + endv[j]);
        #pragma unroll
        for (int off = 16; off > 0; off >>= 1)
            m = fmaxf(m, __shfl_xor(m, off, 32));
        float s = 0.0f;
        for (int j = lane; j < KK; j += 32) s += __expf(alpha[b * KK + j] + endv[j] - m);
        #pragma unroll
        for (int off = 16; off > 0; off >>= 1)
            s += __shfl_xor(s, off, 32);
        if (lane == 0) logz[b0 + b] = m + __logf(s);

        // argmax_j(svit + end), first-occurrence semantics
        float bm = -1e30f; int bj = 0;
        for (int j = lane; j < KK; j += 32) {
            float v = svit[b * KK + j] + endv[j];
            if (v > bm) { bm = v; bj = j; }
        }
        #pragma unroll
        for (int off = 16; off > 0; off >>= 1) {
            float om = __shfl_xor(bm, off, 32);
            int   oj = __shfl_xor(bj, off, 32);
            if (om > bm || (om == bm && oj < bj)) { bm = om; bj = oj; }
        }
        if (lane == 0) last_tag[b0 + b] = bj;
    }
}

// ---------------------------------------------------------------------------
// Kernel 4: backtrack (one thread per batch; paths written as float)
// ---------------------------------------------------------------------------
__global__ void crf_backtrack(const int* __restrict__ last_tag,
                              const unsigned char* __restrict__ bp,
                              float* __restrict__ out) {
    const int b = threadIdx.x;
    float* pout = out + 1;               // out[0] = loss
    int cur = last_tag[b];
    pout[b * TT + (TT - 1)] = (float)cur;
    for (int t = TT - 1; t >= 1; --t) {
        cur = bp[((size_t)b * (TT - 1) + (t - 1)) * KK + cur];
        pout[b * TT + (t - 1)] = (float)cur;
    }
}

// ---------------------------------------------------------------------------
// Kernel 5: loss = -mean(score - log_z)
// ---------------------------------------------------------------------------
__global__ void crf_loss(const float* __restrict__ score,
                         const float* __restrict__ logz,
                         float* __restrict__ out) {
    __shared__ float red[BB];
    const int tid = threadIdx.x;
    red[tid] = score[tid] - logz[tid];
    __syncthreads();
    for (int off = BB / 2; off > 0; off >>= 1) {
        if (tid < off) red[tid] += red[tid + off];
        __syncthreads();
    }
    if (tid == 0) out[0] = -red[0] / (float)BB;
}

// ---------------------------------------------------------------------------
extern "C" void kernel_launch(void* const* d_in, const int* in_sizes, int n_in,
                              void* d_out, int out_size, void* d_ws, size_t ws_size,
                              hipStream_t stream) {
    const float* em       = (const float*)d_in[0];   // (B,T,K) f32
    const float* tr       = (const float*)d_in[1];   // (K,K)   f32
    const float* startv   = (const float*)d_in[2];   // (K,)    f32
    const float* endv     = (const float*)d_in[3];   // (K,)    f32
    const int*   tags     = (const int*)d_in[4];     // (B,T)   i32
    const int*   seq_lens = (const int*)d_in[5];     // (B,)    i32
    float* out = (float*)d_out;

    char* ws = (char*)d_ws;
    _Float16*      Et    = (_Float16*)(ws);            // 32768 B : exp(Tr)^T f16
    float*         score = (float*)(ws + 32768);       //   512 B
    float*         logz  = (float*)(ws + 33280);       //   512 B
    int*           ltag  = (int*)(ws + 33792);         //   512 B
    unsigned char* bp    = (unsigned char*)(ws + 65536); // B*(T-1)*K = 8,372,224 B

    crf_prep_et  <<<KK, KK, 0, stream>>>(tr, Et);
    crf_score    <<<BB, 32, 0, stream>>>(em, tr, startv, endv, tags, seq_lens, score);
    crf_scan     <<<BB / BPG, NTHR, SMEM_BYTES, stream>>>(em, tr, startv, endv,
                                                          seq_lens, Et, logz, ltag, bp);
    crf_backtrack<<<1, BB, 0, stream>>>(ltag, bp, out);
    crf_loss     <<<1, BB, 0, stream>>>(score, logz, out);
}
